// LatentOddityQuantizer_635655160115
// MI455X (gfx1250) — compile-verified, matrix operand order checked by reading, not run on hardware
//
#include <hip/hip_runtime.h>
#include <hip/hip_bf16.h>

// ---- problem constants (B=512,T=16,D=128,K=1024) ----
#define NB   8192      // B*T
#define DIM  128       // D
#define KK   1024      // codebook size
#define DD2  16384     // D*D
#define DECAYF 0.99f
#define ONEMD  0.01f
#define EPSF   1e-5f

typedef __attribute__((ext_vector_type(16))) __bf16 v16bf;
typedef __attribute__((ext_vector_type(8)))  __bf16 v8bf;
typedef __attribute__((ext_vector_type(8)))  float  v8f;

__device__ __forceinline__ v16bf cat16(v8bf lo, v8bf hi) {
  return __builtin_shufflevector(lo, hi, 0,1,2,3,4,5,6,7,8,9,10,11,12,13,14,15);
}
__device__ __forceinline__ v8bf splat8(__bf16 s) {
  v8bf v = {s,s,s,s,s,s,s,s};
  return v;
}

// ---------------------------------------------------------------------------
// Kernel A: embedding fp32 -> bf16 table (256 KB, stays cache-resident)
// ---------------------------------------------------------------------------
__global__ void k_ebf16(const float* __restrict__ emb, __bf16* __restrict__ E) {
  int i = blockIdx.x * 256 + threadIdx.x;
  if (i < KK * DIM) E[i] = (__bf16)emb[i];
}

// ---------------------------------------------------------------------------
// Kernel C: fused  Gz / zGz / distance GEMM / argmin.  One G pass total.
//   workgroup: 256 threads (8 waves) = 16 n-rows x ALL 1024 codes.
//   wave w owns codes [w*128,(w+1)*128): 8 WMMA k-tiles, v8f acc each.
//   G staging is software-pipelined with distance 4 (4 pending float2 regs)
//   so HBM latency is covered by ~32 WMMAs of slack per wave.
// ---------------------------------------------------------------------------
__global__ __launch_bounds__(256)
void k_dist(const float* __restrict__ z, const float* __restrict__ G,
            const __bf16* __restrict__ E,
            float* __restrict__ idx_out_f, int* __restrict__ idx_out_i) {
  __shared__ __align__(16) __bf16 Ast[2][16][32];   // double-buffered A chunk
  __shared__ __align__(16) float zsh[16][DIM];      // z rows (fp32)      8 KB
  __shared__ float u_lds[16][DIM];                  // (G z)              8 KB
  __shared__ float v_lds[16][DIM];                  // (G^T z)            8 KB
  __shared__ __align__(16) __bf16 Zst[16][DIM];     // -(u+v) bf16        4 KB
  __shared__ float zgsh[16];
  __shared__ float Rv[8][8][32];
  __shared__ int   Ri[8][8][32];

  const int tid  = threadIdx.x;
  const int wave = tid >> 5;        // 0..7
  const int lane = tid & 31;
  const int lc   = lane & 15;       // row (A) / code column (B) within tile
  const int lh   = lane >> 4;       // half selector for K packing
  const int n0   = blockIdx.x * 16;
  const int wk0  = wave * 128;

  // ---- stage the 16 z rows straight into LDS with async loads ----
  {
    #pragma unroll
    for (int it = 0; it < 2; ++it) {
      int p  = tid + 256 * it;          // 512 x b128 covers 16x128 fp32
      int r  = p >> 5;
      int c4 = (p & 31) * 4;
      unsigned lo = (unsigned)(uintptr_t)&zsh[r][c4];   // LDS byte offset
      unsigned long long ga =
          (unsigned long long)(uintptr_t)(z + (size_t)(n0 + r) * DIM + c4);
      asm volatile("global_load_async_to_lds_b128 %0, %1, off"
                   :: "v"(lo), "v"(ga) : "memory");
    }
    asm volatile("s_wait_asynccnt 0x0" ::: "memory");
  }
  __syncthreads();

  v8f acc[8] = {};                  // 64 fp32 accumulators / lane

  // staging geometry: 16 threads per n-row, 2 fp32 each per chunk
  const int sr = tid >> 4;          // n-row 0..15
  const int sg = tid & 15;          // slot in row group (contiguous lanes)
  const int sc = sg * 2;            // even column within 32-chunk
  const float* gbase = G + (size_t)(n0 + sr) * DD2 + sc;

  float upart = 0.f;
  float vacc[8] = {};               // constant-indexed after unroll

  // convert + LDS-store one chunk, and fold its values into u / v
  auto put = [&](float2 g, int c, int ccn, int buf) {
    Ast[buf][sr][sc]     = (__bf16)g.x;
    Ast[buf][sr][sc + 1] = (__bf16)g.y;
    const int i  = c >> 2;                  // matrix row inside G_n
    const int c0 = ccn * 32 + sc;           // matrix column of g.x
    upart += g.x * zsh[sr][c0] + g.y * zsh[sr][c0 + 1];
    const float zi = zsh[sr][i];
    vacc[ccn * 2]     += g.x * zi;
    vacc[ccn * 2 + 1] += g.y * zi;
    if (ccn == 3) {                         // row i complete -> reduce u
      float u = upart;
      u += __shfl_xor(u, 1, 32);
      u += __shfl_xor(u, 2, 32);
      u += __shfl_xor(u, 4, 32);
      u += __shfl_xor(u, 8, 32);
      if (sg == 0) u_lds[sr][i] = u;
      upart = 0.f;
    }
  };
  auto lda = [&](int c) -> float2 { return *(const float2*)(gbase + c * 32); };

  // prologue: chunk 0 goes straight to LDS; chunks 1..4 into pending regs
  put(lda(0), 0, 0, 0);
  float2 pend[4];
  pend[1] = lda(1);
  pend[2] = lda(2);
  pend[3] = lda(3);
  pend[0] = lda(4);

  __bf16 s[8];
  for (int cb = 0; cb < 512; cb += 4) {     // 512 chunks of 32 along D*D
    const int i = cb >> 2;                  // embedding row for this i
    #pragma unroll
    for (int cc = 0; cc < 4; ++cc) {
      const int c   = cb + cc;
      const int buf = cc & 1;
      __syncthreads();
      if (cc == 0) {
        #pragma unroll
        for (int t = 0; t < 8; ++t)
          s[t] = E[(size_t)(wk0 + t * 16 + lc) * DIM + i];
      }
      v8bf alo = *(const v8bf*)&Ast[buf][lc][lh * 8];
      v8bf ahi = *(const v8bf*)&Ast[buf][lc][16 + lh * 8];
      v16bf a  = cat16(alo, ahi);
      if (c + 1 < 512) {
        put(pend[(cc + 1) & 3], c + 1, (cc + 1) & 3, buf ^ 1);
        if (c + 5 < 512) pend[(cc + 1) & 3] = lda(c + 5);  // distance-4 load
      }
      const int j0 = cc * 32 + lh * 8;
      #pragma unroll
      for (int t = 0; t < 8; ++t) {
        const __bf16* er = E + (size_t)(wk0 + t * 16 + lc) * DIM;
        v8bf e0 = *(const v8bf*)(er + j0);
        v8bf e1 = *(const v8bf*)(er + j0 + 16);
        v8bf vs = splat8(s[t]);
        acc[t] = __builtin_amdgcn_wmma_f32_16x16x32_bf16(
            false, a, false, cat16(e0 * vs, e1 * vs), (short)0, acc[t],
            false, false);
      }
    }
  }
  __syncthreads();

  // ---- finish Gz: store v, build -(u+v) bf16 tile and zGz ----
  #pragma unroll
  for (int ccn = 0; ccn < 4; ++ccn) {
    v_lds[sr][ccn * 32 + sc]     = vacc[ccn * 2];
    v_lds[sr][ccn * 32 + sc + 1] = vacc[ccn * 2 + 1];
  }
  __syncthreads();
  for (int p = tid; p < 16 * DIM; p += 256) {
    int r = p >> 7, cx = p & 127;
    Zst[r][cx] = (__bf16)(-(u_lds[r][cx] + v_lds[r][cx]));   // == -2*Gz
  }
  if (tid < 16) {
    float zg = 0.f;
    for (int cx = 0; cx < DIM; ++cx)
      zg += 0.5f * (u_lds[tid][cx] + v_lds[tid][cx]) * zsh[tid][cx];
    zgsh[tid] = zg;                                           // z.Gs.z
  }
  __syncthreads();

  // ---- fold in -2*eGz with 4 more WMMA chunks against E ----
  #pragma unroll
  for (int cc = 0; cc < 4; ++cc) {
    const int j0 = cc * 32 + lh * 8;
    v8bf alo = *(const v8bf*)&Zst[lc][j0];
    v8bf ahi = *(const v8bf*)&Zst[lc][j0 + 16];
    v16bf a  = cat16(alo, ahi);
    #pragma unroll
    for (int t = 0; t < 8; ++t) {
      const __bf16* er = E + (size_t)(wk0 + t * 16 + lc) * DIM;
      v8bf e0 = *(const v8bf*)(er + j0);
      v8bf e1 = *(const v8bf*)(er + j0 + 16);
      acc[t] = __builtin_amdgcn_wmma_f32_16x16x32_bf16(
          false, a, false, cat16(e0, e1), (short)0, acc[t], false, false);
    }
  }

  // ---- dist = acc + zGz ; per-lane argmin over this wave's 128 codes ----
  const int m0 = (lane < 16) ? 0 : 8;      // C/D layout: VGPR r -> row r(+8)
  #pragma unroll
  for (int r = 0; r < 8; ++r) {
    float best = 3.4e38f; int bk = 0;
    const float zg = zgsh[r + m0];
    #pragma unroll
    for (int t = 0; t < 8; ++t) {
      float dv = acc[t][r] + zg;
      int   k  = wk0 + t * 16 + lc;        // k increases with t -> '<' keeps lowest
      if (dv < best) { best = dv; bk = k; }
    }
    Rv[wave][r][lane] = best;
    Ri[wave][r][lane] = bk;
  }
  __syncthreads();

  // ---- final argmin across 8 waves x 16 lanes per row ----
  if (tid < 16) {
    const int m = tid;
    const int r = m & 7;
    const int lb = (m < 8) ? 0 : 16;
    float best = 3.4e38f; int bk = 0x7fffffff;
    for (int w = 0; w < 8; ++w)
      for (int l2 = 0; l2 < 16; ++l2) {
        float v = Rv[w][r][lb + l2];
        int   k = Ri[w][r][lb + l2];
        if (v < best || (v == best && k < bk)) { best = v; bk = k; }
      }
    idx_out_f[n0 + m] = (float)bk;
    idx_out_i[n0 + m] = bk;
  }
}

// ---------------------------------------------------------------------------
// Kernel D1: deterministic per-code EMA (segment sums in fixed n-order).
// one 128-thread block per code k.
// ---------------------------------------------------------------------------
__global__ void k_ema(const float* __restrict__ z, const int* __restrict__ idx,
                      const float* __restrict__ cs, const float* __restrict__ emaw,
                      float* __restrict__ out_ncs, float* __restrict__ out_nema) {
  __shared__ int ish[DIM];
  const int k = blockIdx.x;
  const int d = threadIdx.x;
  float acc = 0.f;
  int cnt = 0;
  for (int base = 0; base < NB; base += DIM) {
    ish[d] = idx[base + d];
    __syncthreads();
    for (int q = 0; q < DIM; ++q) {
      if (ish[q] == k) { acc += z[(size_t)(base + q) * DIM + d]; ++cnt; }
    }
    __syncthreads();
  }
  out_nema[(size_t)k * DIM + d] = emaw[(size_t)k * DIM + d] * DECAYF + ONEMD * acc;
  if (d == 0) out_ncs[k] = cs[k] * DECAYF + ONEMD * (float)cnt;
}

// ---------------------------------------------------------------------------
// Kernel D2: n = sum(ncs); smoothed; new_embedding = nema / smoothed
// ---------------------------------------------------------------------------
__global__ void k_final(const float* __restrict__ out_ncs,
                        const float* __restrict__ out_nema,
                        float* __restrict__ out_emb) {
  __shared__ float red[256];
  __shared__ float nsum;
  const int t = threadIdx.x;
  float s = 0.f;
  for (int k = t; k < KK; k += 256) s += out_ncs[k];
  red[t] = s;
  __syncthreads();
  for (int st = 128; st > 0; st >>= 1) {
    if (t < st) red[t] += red[t + st];
    __syncthreads();
  }
  if (t == 0) nsum = red[0];
  __syncthreads();
  const float n = nsum;
  const float scale = n / (n + (float)KK * EPSF);
  for (int p = t; p < KK * DIM; p += 256) {
    int k = p >> 7;
    float smoothed = (out_ncs[k] + EPSF) * scale;
    out_emb[p] = out_nema[p] / smoothed;
  }
}

// ---------------------------------------------------------------------------
extern "C" void kernel_launch(void* const* d_in, const int* in_sizes, int n_in,
                              void* d_out, int out_size, void* d_ws, size_t ws_size,
                              hipStream_t stream) {
  (void)in_sizes; (void)n_in; (void)out_size; (void)ws_size;
  const float* z    = (const float*)d_in[0];   // [B,T,D]
  const float* G    = (const float*)d_in[1];   // [B,T,D,D]
  const float* emb  = (const float*)d_in[2];   // [K,D]
  const float* cs   = (const float*)d_in[3];   // [K]
  const float* emaw = (const float*)d_in[4];   // [K,D]

  // workspace carve-out (~300 KB)
  char* ws = (char*)d_ws;
  __bf16* E   = (__bf16*)ws;                   // K*D bf16 = 256 KB
  int*    iwk = (int*)(ws + 262144);           // N int

  float* out    = (float*)d_out;
  float* o_idx  = out;                         // N
  float* o_emb  = out + NB;                    // K*D
  float* o_ncs  = out + NB + KK * DIM;         // K
  float* o_nema = out + NB + KK * DIM + KK;    // K*D

  k_ebf16<<<(KK * DIM + 255) / 256, 256, 0, stream>>>(emb, E);
  k_dist <<<NB / 16, 256, 0, stream>>>(z, G, E, o_idx, iwk);
  k_ema  <<<KK, DIM, 0, stream>>>(z, iwk, cs, emaw, o_ncs, o_nema);
  k_final<<<1, 256, 0, stream>>>(o_ncs, o_nema, o_emb);
}
